// TemporalSelfAttention_73504070303828
// MI455X (gfx1250) — compile-verified, matrix operand order checked by reading, not run on hardware
//
#include <hip/hip_runtime.h>
#include <hip/hip_bf16.h>

typedef __attribute__((ext_vector_type(16))) _Float16 v16h;
typedef __attribute__((ext_vector_type(8)))  _Float16 v8h;
typedef __attribute__((ext_vector_type(2)))  _Float16 v2h;
typedef __attribute__((ext_vector_type(8)))  float    v8f;
typedef __attribute__((ext_vector_type(4)))  unsigned int u32x4;
typedef __attribute__((ext_vector_type(8)))  unsigned int u32x8;

#define BATCH 256
#define NPOS  512
#define DIM   64
#define XS    72          // row stride (halves) for X / K / Q-scratch (bank-conflict free)
#define WS    72          // row stride (halves) for transposed weights
#define VS    520         // row stride (halves) for Vt [e][m']
#define QSCR  (16 * XS)   // per-wave 16x64 f16 scratch (Q / O round-trip)
#define PSTR  40
#define PSCR  (16 * PSTR) // per-wave 16x32 f16 P-tile scratch

// Load a 16x32 f16 A-style fragment (also used for B as B^T) from LDS,
// row-major with given stride. Matches CDNA5 16-bit A 16x32 VGPR layout:
// lane<16: row=lane, K=0..7 & 16..23 ; lane>=16: row=lane-16, K=8..15 & 24..31.
__device__ __forceinline__ v16h frag_ld(const _Float16* p, int stride) {
  const int lane = threadIdx.x & 31;
  const int row  = lane & 15;
  const int koff = (lane < 16) ? 0 : 8;
  const _Float16* q = p + row * stride + koff;
  v8h lo = *(const v8h*)q;          // 16B aligned (strides are multiples of 8 halves)
  v8h hi = *(const v8h*)(q + 16);
  v16h r;
#pragma unroll
  for (int i = 0; i < 8; ++i) { r[i] = lo[i]; r[i + 8] = hi[i]; }
  return r;
}

__device__ __forceinline__ v8f wmma_f16(v16h a, v16h b, v8f c) {
  return __builtin_amdgcn_wmma_f32_16x16x32_f16(false, a, false, b, (short)0, c,
                                                false, false);
}

// Store a 16x16 f32 C/D tile (lane=col n within half, VGPR r -> row 8*half+r)
// to row-major f16 LDS with stride, applying a scale.
__device__ __forceinline__ void tile_st(_Float16* p, int stride, v8f acc, float scale) {
  const int lane = threadIdx.x & 31;
  const int col  = lane & 15;
  const int r0   = (lane >> 4) * 8;
#pragma unroll
  for (int r = 0; r < 8; ++r)
    p[(r0 + r) * stride + col] = (_Float16)(acc[r] * scale);
}

__global__ __launch_bounds__(256) void attn_branch_kernel(
    const float* __restrict__ xg, const float* __restrict__ wq,
    const float* __restrict__ wk, const float* __restrict__ wv,
    const float* __restrict__ wd, const float* __restrict__ bd,
    float* __restrict__ out, int mode) {
  extern __shared__ _Float16 smem[];
  _Float16* Xs   = smem;                   // [512][72]
  _Float16* Ks   = Xs + NPOS * XS;         // [512][72]
  _Float16* Vt   = Ks + NPOS * XS;         // [64][520]  V^T with interleaved m'
  _Float16* WqT  = Vt + DIM * VS;          // [64][72]   W^T: WT[e][d]
  _Float16* WkT  = WqT + DIM * WS;
  _Float16* WvT  = WkT + DIM * WS;
  _Float16* WdT  = WvT + DIM * WS;         // d-rows permuted (interleave within 32)
  _Float16* Qscr = WdT + DIM * WS;         // 8 waves x [16][72]
  _Float16* Pscr = Qscr + 8 * QSCR;        // 8 waves x [16][40]

  const int b    = blockIdx.x;
  const int tid  = threadIdx.x;
  const int wave = tid >> 5;
  const int lane = tid & 31;
  const int col  = lane & 15;
  const int r0   = (lane >> 4) * 8;
  const float* xb = xg + (size_t)b * NPOS * DIM;

  // f32 staging area for the TDM copy: reuse Ks+Vt region (140 KB >= 128 KB).
  float* stage = (float*)Ks;

  // ---------------- Phase 0: TDM DMA of x[b] (512x64 f32) into LDS ----------
  if (wave == 0) {
    unsigned long long ga = (unsigned long long)(void*)xb;
    unsigned ldsoff = (unsigned)(unsigned long long)(void*)stage; // low 32b = LDS offset
    u32x4 g0;
    g0[0] = 1u;                                    // count=1, user descriptor
    g0[1] = ldsoff;                                // lds_addr
    g0[2] = (unsigned)ga;                          // global_addr[31:0]
    g0[3] = (unsigned)((ga >> 32) & 0x1FFFFFFull)  // global_addr[56:32]
            | (2u << 30);                          // type=2 ("image")
    u32x8 g1;
    g1[0] = 2u << 16;                              // data_size=2 (4 bytes)
    g1[1] = (unsigned)DIM << 16;                   // tensor_dim0 = 64 (low 16)
    g1[2] = (unsigned)NPOS << 16;                  // dim0 hi=0 | tensor_dim1=512 lo
    g1[3] = (unsigned)DIM << 16;                   // dim1 hi=0 | tile_dim0=64
    g1[4] = (unsigned)NPOS;                        // tile_dim1=512 | tile_dim2=0
    g1[5] = (unsigned)DIM;                         // tensor_dim0_stride = 64
    g1[6] = 0u;
    g1[7] = 0u;
    asm volatile("tensor_load_to_lds %0, %1" :: "s"(g0), "s"(g1) : "memory");
    __builtin_amdgcn_s_wait_tensorcnt(0);
  }
  __syncthreads();

  // ---------------- Phase 1: convert staged f32 -> f16, stage weights -------
  for (int i = tid; i < (NPOS * DIM) / 4; i += 256) {
    float4 f = ((const float4*)stage)[i];
    int n = (i << 2) >> 6;
    int d = (i << 2) & 63;
    _Float16* dst = &Xs[n * XS + d];
    dst[0] = (_Float16)f.x; dst[1] = (_Float16)f.y;
    dst[2] = (_Float16)f.z; dst[3] = (_Float16)f.w;
  }
  for (int i = tid; i < DIM * DIM; i += 256) {
    int d = i >> 6, e = i & 63;
    WqT[e * WS + d] = (_Float16)wq[i];
    WkT[e * WS + d] = (_Float16)wk[i];
    WvT[e * WS + d] = (_Float16)wv[i];
    // WdT d-rows permuted to match interleaved O round-trip columns:
    // dp = (d & 32) | 2*(d % 16) | ((d>>4) & 1)
    int dp = (d & 32) | ((d & 15) << 1) | ((d >> 4) & 1);
    WdT[e * WS + dp] = (_Float16)wd[i];
  }
  __syncthreads();

  // ---------------- Phase 2: K = X*Wk (row-major), V = X*Wv (transposed) -----
  for (int t = wave; t < 128; t += 8) {      // 32 m-blocks x 4 e-blocks
    int mb = t >> 2, eb = t & 3;
    v8f aK = {}, aV = {};
#pragma unroll
    for (int kc = 0; kc < 2; ++kc) {
      v16h a = frag_ld(&Xs[(mb * 16) * XS + kc * 32], XS);
      aK = wmma_f16(a, frag_ld(&WkT[(eb * 16) * WS + kc * 32], WS), aK);
      aV = wmma_f16(a, frag_ld(&WvT[(eb * 16) * WS + kc * 32], WS), aV);
    }
    tile_st(&Ks[(mb * 16) * XS + eb * 16], XS, aK, 1.0f);
    // Vt[e][m'] with m interleaved within each 32-block:
    // m = mb*16 + r0 + r ; m' = (mb/2)*32 + 2*(r0+r) + (mb&1)
#pragma unroll
    for (int r = 0; r < 8; ++r)
      Vt[(eb * 16 + col) * VS + (mb >> 1) * 32 + ((r0 + r) << 1) + (mb & 1)] =
          (_Float16)aV[r];
  }
  __syncthreads();

  // ---------------- Phase 3: per-wave flash attention (4 q-blocks each) ------
  _Float16* myQ = Qscr + wave * QSCR;
  _Float16* myP = Pscr + wave * PSCR;

  v16h ones;                                  // J fragment: rowsum(P) = P x J
#pragma unroll
  for (int i = 0; i < 16; ++i) ones[i] = (_Float16)1.0f;

  for (int qi = 0; qi < 4; ++qi) {
    const int qb = wave * 4 + qi;            // q-block 0..31 (16 rows)

    // Q = Xblk * Wq, scale 1/sqrt(64) folded into f16 conversion
#pragma unroll
    for (int eb = 0; eb < 4; ++eb) {
      v8f acc = {};
#pragma unroll
      for (int kc = 0; kc < 2; ++kc)
        acc = wmma_f16(frag_ld(&Xs[(qb * 16) * XS + kc * 32], XS),
                       frag_ld(&WqT[(eb * 16) * WS + kc * 32], WS), acc);
      tile_st(&myQ[eb * 16], XS, acc, 0.125f);
    }
    v16h qf0 = frag_ld(&myQ[0],  XS);
    v16h qf1 = frag_ld(&myQ[32], XS);

    float rmax[8], rsum[8];
#pragma unroll
    for (int r = 0; r < 8; ++r) { rmax[r] = -1e30f; rsum[r] = 0.0f; }
    v8f o0 = {}, o1 = {}, o2 = {}, o3 = {};

    for (int mp = 0; mp < 16; ++mp) {        // 32 K-rows per iteration
      v8f s0 = {}, s1 = {};
      s0 = wmma_f16(qf0, frag_ld(&Ks[(mp * 32) * XS + 0],  XS), s0);
      s0 = wmma_f16(qf1, frag_ld(&Ks[(mp * 32) * XS + 32], XS), s0);
      s1 = wmma_f16(qf0, frag_ld(&Ks[(mp * 32 + 16) * XS + 0],  XS), s1);
      s1 = wmma_f16(qf1, frag_ld(&Ks[(mp * 32 + 16) * XS + 32], XS), s1);

      // Online softmax: only the row-max needs lane shuffles (rows live in one
      // 16-lane half -> xor 1,2,4,8). Row-sum is done on the matrix unit below.
      float cc[8];
#pragma unroll
      for (int r = 0; r < 8; ++r) {
        float mnew = fmaxf(s0[r], s1[r]);
#pragma unroll
        for (int off = 1; off < 16; off <<= 1)
          mnew = fmaxf(mnew, __shfl_xor(mnew, off, 32));
        float mi = fmaxf(rmax[r], mnew);
        float c  = __expf(rmax[r] - mi);
        rmax[r]  = mi;
        cc[r]    = c;
        float p0 = __expf(s0[r] - mi);
        float p1 = __expf(s1[r] - mi);
        o0[r] *= c; o1[r] *= c; o2[r] *= c; o3[r] *= c;
        // Interleaved m': (p0,p1) -> columns 2*col, 2*col+1 (single b32 store).
        v2h pp; pp[0] = (_Float16)p0; pp[1] = (_Float16)p1;
        *(v2h*)&myP[(r0 + r) * PSTR + (col << 1)] = pp;
      }
      v16h pf = frag_ld(myP, PSTR);          // P tile back as A-fragment
      v8f sacc = {};
      sacc = wmma_f16(pf, ones, sacc);       // rowsum(P) on the matrix unit
      o0 = wmma_f16(pf, frag_ld(&Vt[(0 * 16) * VS + mp * 32], VS), o0);
      o1 = wmma_f16(pf, frag_ld(&Vt[(1 * 16) * VS + mp * 32], VS), o1);
      o2 = wmma_f16(pf, frag_ld(&Vt[(2 * 16) * VS + mp * 32], VS), o2);
      o3 = wmma_f16(pf, frag_ld(&Vt[(3 * 16) * VS + mp * 32], VS), o3);
#pragma unroll
      for (int r = 0; r < 8; ++r) rsum[r] = rsum[r] * cc[r] + sacc[r];
    }

    // Normalize rows; pack (o0,o1)/(o2,o3) into interleaved columns matching
    // the WdT d-row permutation, then round-trip through scratch.
#pragma unroll
    for (int r = 0; r < 8; ++r) {
      float inv = 1.0f / rsum[r];
      v2h a; a[0] = (_Float16)(o0[r] * inv); a[1] = (_Float16)(o1[r] * inv);
      v2h c2; c2[0] = (_Float16)(o2[r] * inv); c2[1] = (_Float16)(o3[r] * inv);
      *(v2h*)&myQ[(r0 + r) * XS + (col << 1)]      = a;
      *(v2h*)&myQ[(r0 + r) * XS + 32 + (col << 1)] = c2;
    }
    v16h of0 = frag_ld(&myQ[0],  XS);
    v16h of1 = frag_ld(&myQ[32], XS);

#pragma unroll
    for (int eb = 0; eb < 4; ++eb) {
      v8f y = {};
      y = wmma_f16(of0, frag_ld(&WdT[(eb * 16) * WS + 0],  WS), y);
      y = wmma_f16(of1, frag_ld(&WdT[(eb * 16) * WS + 32], WS), y);
      float bias  = bd[eb * 16 + col];
      size_t base = (size_t)b * NPOS * DIM + (size_t)(qb * 16 + r0) * DIM + eb * 16 + col;
#pragma unroll
      for (int r = 0; r < 8; ++r) {
        size_t idx = base + (size_t)r * DIM;
        float yv = y[r] + bias;
        if (mode == 0) {
          out[idx] = yv;                                  // stash temporal Yt
        } else {
          float yt = out[idx];                            // read Yt back
          float g  = 1.0f / (1.0f + __expf(-(yt + yv)));  // sigmoid(Yt+Ys)
          out[idx] = xb[(size_t)(qb * 16 + r0 + r) * DIM + eb * 16 + col] * g;
        }
      }
    }
  }
}

extern "C" void kernel_launch(void* const* d_in, const int* in_sizes, int n_in,
                              void* d_out, int out_size, void* d_ws, size_t ws_size,
                              hipStream_t stream) {
  const float* x    = (const float*)d_in[0];
  const float* wq_t = (const float*)d_in[1];
  const float* wk_t = (const float*)d_in[2];
  const float* wv_t = (const float*)d_in[3];
  const float* wq_s = (const float*)d_in[4];
  const float* wk_s = (const float*)d_in[5];
  const float* wv_s = (const float*)d_in[6];
  const float* wd_t = (const float*)d_in[7];
  const float* bd_t = (const float*)d_in[8];
  const float* wd_s = (const float*)d_in[9];
  const float* bd_s = (const float*)d_in[10];
  float* out = (float*)d_out;
  (void)d_ws; (void)ws_size; (void)in_sizes; (void)n_in; (void)out_size;

  const size_t shbytes =
      (size_t)(2 * NPOS * XS + DIM * VS + 4 * DIM * WS + 8 * QSCR + 8 * PSCR) *
      sizeof(_Float16);  // ~273 KB of the 320 KB WGP LDS

  // Pass 1: temporal branch -> Yt stashed in d_out.
  attn_branch_kernel<<<BATCH, 256, shbytes, stream>>>(x, wq_t, wk_t, wv_t, wd_t,
                                                      bd_t, out, 0);
  // Pass 2: spatial branch, then out = x * sigmoid(Yt + Ys).
  attn_branch_kernel<<<BATCH, 256, shbytes, stream>>>(x, wq_s, wk_s, wv_s, wd_s,
                                                      bd_s, out, 1);
}
